// DenseReparam_25804163514601
// MI455X (gfx1250) — matrix operand order repaired and checked by reading, not compiled, not run on hardware
//
#include <hip/hip_runtime.h>
#include <stdint.h>

#define BATCH 8192
#define NIN   4096
#define UNITS 4096

#define BM 128
#define BN 128
#define BK 32
#define LDK 40   // padded LDS row stride in ushort units (80B = 5*16B: aligned, conflict-free b128 frag reads)
#define NBUF 4   // async pipeline buffers (4 x 40KB = 160KB of the WGP's 320KB LDS); index = kt & 3

typedef __attribute__((ext_vector_type(16))) __bf16        bf16x16;
typedef __attribute__((ext_vector_type(8)))  float         floatx8;
typedef __attribute__((ext_vector_type(4)))  float         float4v;
typedef __attribute__((ext_vector_type(4)))  unsigned int  uint4v;
typedef __attribute__((ext_vector_type(2)))  unsigned int  uint2v;

union FragU { bf16x16 v; uint4v q[2]; };

__device__ __forceinline__ unsigned short f2bf(float f) {
    unsigned u = __float_as_uint(f);
    u += 0x7FFFu + ((u >> 16) & 1u);   // round-to-nearest-even to bf16
    return (unsigned short)(u >> 16);
}
__device__ __forceinline__ float bf2f(unsigned short h) {
    return __uint_as_float(((unsigned)h) << 16);
}

// ---------------------------------------------------------------------------
// Pass 1 of hyperspherical->cartesian: per (column, segment) product of sins.
// ---------------------------------------------------------------------------
__global__ void hs2c_segprod(const float* __restrict__ theta, float* __restrict__ P) {
    int t   = blockIdx.x * blockDim.x + threadIdx.x;  // 0 .. 32*4096-1
    int seg = t >> 12;
    int col = t & (UNITS - 1);
    int r0  = seg * 128;
    float p = 1.0f;
    #pragma unroll 4
    for (int i = 0; i < 128; ++i) {
        int row = r0 + i;
        if (row < NIN - 1) p *= sinf(theta[(size_t)row * UNITS + col]);
    }
    P[seg * UNITS + col] = p;
}

// ---------------------------------------------------------------------------
// Pass 2: prefix over segment products, then emit v rows for this segment,
// written TRANSPOSED (Vt[n][k] = v[k][n]) and pre-split into bf16 hi/lo.
// ---------------------------------------------------------------------------
__global__ void hs2c_buildv(const float* __restrict__ theta, const float* __restrict__ P,
                            unsigned short* __restrict__ Vth, unsigned short* __restrict__ Vtl) {
    int t   = blockIdx.x * blockDim.x + threadIdx.x;
    int seg = t >> 12;
    int col = t & (UNITS - 1);
    float run = 1.0f;
    #pragma unroll
    for (int s = 0; s < 32; ++s)
        if (s < seg) run *= P[s * UNITS + col];
    int r0 = seg * 128;
    for (int i0 = 0; i0 < 128; i0 += 8) {
        unsigned short h8[8], l8[8];
        #pragma unroll
        for (int j = 0; j < 8; ++j) {
            int row = r0 + i0 + j;
            float val;
            float s = 1.0f;
            if (row < NIN - 1) {
                float a, c;
                a = theta[(size_t)row * UNITS + col];
                sincosf(a, &s, &c);
                val = c * run;       // v[row] = cos * prod(sin before)
            } else {
                val = run;           // row 4095: v = full sin product
            }
            h8[j] = f2bf(val);
            l8[j] = f2bf(val - bf2f(h8[j]));
            if (row < NIN - 1) run *= s;
        }
        uint4v ph = {(unsigned)h8[0] | ((unsigned)h8[1] << 16),
                     (unsigned)h8[2] | ((unsigned)h8[3] << 16),
                     (unsigned)h8[4] | ((unsigned)h8[5] << 16),
                     (unsigned)h8[6] | ((unsigned)h8[7] << 16)};
        uint4v pl = {(unsigned)l8[0] | ((unsigned)l8[1] << 16),
                     (unsigned)l8[2] | ((unsigned)l8[3] << 16),
                     (unsigned)l8[4] | ((unsigned)l8[5] << 16),
                     (unsigned)l8[6] | ((unsigned)l8[7] << 16)};
        *(uint4v*)&Vth[(size_t)col * NIN + r0 + i0] = ph;
        *(uint4v*)&Vtl[(size_t)col * NIN + r0 + i0] = pl;
    }
}

// ---------------------------------------------------------------------------
// One streaming pass: split X (fp32) into bf16 hi/lo arrays.
// ---------------------------------------------------------------------------
__global__ void split_x(const float* __restrict__ X,
                        unsigned short* __restrict__ Xh, unsigned short* __restrict__ Xl) {
    size_t t = (size_t)blockIdx.x * blockDim.x + threadIdx.x;
    float4v f = *(const float4v*)(X + t * 4);
    unsigned short h[4], l[4];
    #pragma unroll
    for (int j = 0; j < 4; ++j) {
        h[j] = f2bf(f[j]);
        l[j] = f2bf(f[j] - bf2f(h[j]));
    }
    uint2v ph = {(unsigned)h[0] | ((unsigned)h[1] << 16),
                 (unsigned)h[2] | ((unsigned)h[3] << 16)};
    uint2v pl = {(unsigned)l[0] | ((unsigned)l[1] << 16),
                 (unsigned)l[2] | ((unsigned)l[3] << 16)};
    *(uint2v*)(Xh + t * 4) = ph;
    *(uint2v*)(Xl + t * 4) = pl;
}

// ---------------------------------------------------------------------------
// Async-copy one 128x32 bf16 tile into padded LDS: GVS mode (uniform SGPR
// base + loop-invariant 32-bit per-lane offset). 2 x b128 per thread.
// ---------------------------------------------------------------------------
__device__ __forceinline__ void async_pair(unsigned long long gbase,
                                           unsigned goff0, unsigned goff1,
                                           unsigned lbase,
                                           unsigned loff0, unsigned loff1) {
    asm volatile("global_load_async_to_lds_b128 %0, %1, %2"
                 :: "v"(lbase + loff0), "v"(goff0), "s"(gbase) : "memory");
    asm volatile("global_load_async_to_lds_b128 %0, %1, %2"
                 :: "v"(lbase + loff1), "v"(goff1), "s"(gbase) : "memory");
}

// ---------------------------------------------------------------------------
// Fused GEMM + epilogue:  out = r * relu(x @ v + lam)
// fp32 emulated via bf16x3 split on V_WMMA_F32_16X16X32_BF16.
// Tiles staged global->LDS by the gfx1250 async engine, distance-3 pipeline.
// ---------------------------------------------------------------------------
__global__ __launch_bounds__(256) void dense_reparam_gemm(
    const unsigned short* __restrict__ Xh, const unsigned short* __restrict__ Xl,
    const unsigned short* __restrict__ Vth, const unsigned short* __restrict__ Vtl,
    const float* __restrict__ theta, float* __restrict__ out) {

    __shared__ __align__(16) unsigned short Ah[NBUF][BM * LDK];
    __shared__ __align__(16) unsigned short Al[NBUF][BM * LDK];
    __shared__ __align__(16) unsigned short Bh[NBUF][BN * LDK];
    __shared__ __align__(16) unsigned short Bl[NBUF][BN * LDK];

    const int tid  = threadIdx.x;
    const int lane = tid & 31;
    const int wave = tid >> 5;
    const int ll   = lane & 15;
    const int lh   = lane >> 4;
    const int wM   = (wave & 1) * 64;   // 2 waves across M (64 rows each)
    const int wN   = (wave >> 1) * 32;  // 4 waves across N (32 cols each)
    const int mB   = blockIdx.y * BM;
    const int nB   = blockIdx.x * BN;

    // Per-thread copy geometry (identical for all four tiles): chunk c covers
    // tile row c>>2, 16B chunk (c&3) of its 64B row.
    const int c0 = tid, c1 = tid + 256;
    const unsigned goff0 = (unsigned)(((c0 >> 2) * NIN + (c0 & 3) * 8) * 2);
    const unsigned goff1 = (unsigned)(((c1 >> 2) * NIN + (c1 & 3) * 8) * 2);
    const unsigned loff0 = (unsigned)(((c0 >> 2) * LDK + (c0 & 3) * 8) * 2);
    const unsigned loff1 = (unsigned)(((c1 >> 2) * LDK + (c1 & 3) * 8) * 2);

    const size_t aRow = (size_t)mB * NIN;
    const size_t bRow = (size_t)nB * NIN;

    auto stage = [&](int kt) {
        const int buf = kt & (NBUF - 1);
        const size_t k0 = (size_t)kt * BK;
        async_pair((unsigned long long)(uintptr_t)(Xh + aRow + k0), goff0, goff1,
                   (unsigned)(uintptr_t)&Ah[buf][0], loff0, loff1);
        async_pair((unsigned long long)(uintptr_t)(Xl + aRow + k0), goff0, goff1,
                   (unsigned)(uintptr_t)&Al[buf][0], loff0, loff1);
        async_pair((unsigned long long)(uintptr_t)(Vth + bRow + k0), goff0, goff1,
                   (unsigned)(uintptr_t)&Bh[buf][0], loff0, loff1);
        async_pair((unsigned long long)(uintptr_t)(Vtl + bRow + k0), goff0, goff1,
                   (unsigned)(uintptr_t)&Bl[buf][0], loff0, loff1);
    };

    floatx8 acc[4][2];
    const floatx8 zero = {0.f, 0.f, 0.f, 0.f, 0.f, 0.f, 0.f, 0.f};
    #pragma unroll
    for (int m = 0; m < 4; ++m)
        #pragma unroll
        for (int n = 0; n < 2; ++n) acc[m][n] = zero;

    auto compute = [&](int kt) {
        const int p = kt & (NBUF - 1);
        FragU bh[2], bl[2];
        #pragma unroll
        for (int n = 0; n < 2; ++n) {
            int base = (wN + n * 16 + ll) * LDK + lh * 8;
            bh[n].q[0] = *(const uint4v*)&Bh[p][base];
            bh[n].q[1] = *(const uint4v*)&Bh[p][base + 16];
            bl[n].q[0] = *(const uint4v*)&Bl[p][base];
            bl[n].q[1] = *(const uint4v*)&Bl[p][base + 16];
        }
        #pragma unroll
        for (int m = 0; m < 4; ++m) {
            int abase = (wM + m * 16 + ll) * LDK + lh * 8;
            FragU ah, al;
            ah.q[0] = *(const uint4v*)&Ah[p][abase];
            ah.q[1] = *(const uint4v*)&Ah[p][abase + 16];
            al.q[0] = *(const uint4v*)&Al[p][abase];
            al.q[1] = *(const uint4v*)&Al[p][abase + 16];
            #pragma unroll
            for (int n = 0; n < 2; ++n) {
                acc[m][n] = __builtin_amdgcn_wmma_f32_16x16x32_bf16(
                    false, ah.v, false, bh[n].v, (short)0, acc[m][n], false, false);
                acc[m][n] = __builtin_amdgcn_wmma_f32_16x16x32_bf16(
                    false, ah.v, false, bl[n].v, (short)0, acc[m][n], false, false);
                acc[m][n] = __builtin_amdgcn_wmma_f32_16x16x32_bf16(
                    false, al.v, false, bh[n].v, (short)0, acc[m][n], false, false);
            }
        }
    };

    const int KT = NIN / BK;  // 128 k-steps

    // Prologue: fill pipeline 3 deep; set 0 must have landed (24 outstanding,
    // allow the 16 newest to remain).
    stage(0);
    stage(1);
    stage(2);
    asm volatile("s_wait_asynccnt 0x10" ::: "memory");
    __syncthreads();

    // Steady state: branch-free body, constant wait count. After staging kt+3,
    // outstanding = sets kt+1,kt+2,kt+3 (24 ops); <=16 ensures set kt+1 done.
    for (int kt = 0; kt < KT - 3; ++kt) {
        stage(kt + 3);
        compute(kt);
        asm volatile("s_wait_asynccnt 0x10" ::: "memory");
        __syncthreads();
    }
    // Drain tail: everything already staged.
    for (int kt = KT - 3; kt < KT; ++kt) {
        compute(kt);
        asm volatile("s_wait_asynccnt 0x0" ::: "memory");
        __syncthreads();
    }

    // Epilogue: out = r * relu(acc + lam); C/D layout (vgpr g, lane l):
    // M = 8*(l/16) + g, N = l%16.
    #pragma unroll
    for (int n = 0; n < 2; ++n) {
        const int   N   = nB + wN + n * 16 + ll;
        const float lam = theta[(size_t)(NIN - 1) * UNITS + N];  // row 4095
        const float rr  = theta[(size_t)NIN * UNITS + N];        // row 4096
        #pragma unroll
        for (int m = 0; m < 4; ++m) {
            const int Mb = mB + wM + m * 16 + lh * 8;
            #pragma unroll
            for (int g = 0; g < 8; ++g) {
                float z = acc[m][n][g] + lam;
                out[(size_t)(Mb + g) * UNITS + N] = rr * fmaxf(z, 0.f);
            }
        }
    }
}

extern "C" void kernel_launch(void* const* d_in, const int* in_sizes, int n_in,
                              void* d_out, int out_size, void* d_ws, size_t ws_size,
                              hipStream_t stream) {
    const float* x     = (const float*)d_in[0];
    const float* theta = (const float*)d_in[1];
    float* out = (float*)d_out;

    // Workspace: Vt_hi (32MB) | Vt_lo (32MB) | X_hi (64MB) | X_lo (64MB) | P (512KB)
    unsigned short* Vth = (unsigned short*)d_ws;
    unsigned short* Vtl = Vth + (size_t)UNITS * NIN;
    unsigned short* Xh  = Vtl + (size_t)UNITS * NIN;
    unsigned short* Xl  = Xh + (size_t)BATCH * NIN;
    float*          P   = (float*)(Xl + (size_t)BATCH * NIN);

    const int scan_threads = 32 * UNITS;
    hs2c_segprod<<<dim3(scan_threads / 256), dim3(256), 0, stream>>>(theta, P);
    hs2c_buildv <<<dim3(scan_threads / 256), dim3(256), 0, stream>>>(theta, P, Vth, Vtl);
    split_x<<<dim3((unsigned)((size_t)BATCH * NIN / 4 / 256)), dim3(256), 0, stream>>>(x, Xh, Xl);

    dim3 grid(UNITS / BN, BATCH / BM);
    dense_reparam_gemm<<<grid, dim3(256), 0, stream>>>(Xh, Xl, Vth, Vtl, theta, out);
}
